// PASent_14310831030952
// MI455X (gfx1250) — matrix operand order repaired and checked by reading, not compiled, not run on hardware
//
#include <hip/hip_runtime.h>
#include <hip/hip_bf16.h>

#define D_DIM   768
#define SEQ     512
#define LC      8
#define CN      128
#define BATCH   32
#define CP      2                 // chunks per workgroup
#define MROWS   (CP * LC)         // 16 rows = one WMMA M tile
#define NCHUNKS (BATCH * CN)      // 4096
#define KSTEPS  (D_DIM / 32)      // 24 WMMA k-steps
#define XPADB   (D_DIM + 8)       // LDS bf16 row stride (shorts) -> conflict-free ds_load_b128
#define XPADF   (D_DIM + 4)       // LDS f32 row stride

typedef __attribute__((ext_vector_type(16))) __bf16 v16bf;
typedef __attribute__((ext_vector_type(8)))  __bf16 v8bf;
typedef __attribute__((ext_vector_type(8)))  float  v8f;

static __device__ __forceinline__ unsigned short f32_to_bf16_rne(float f) {
    unsigned int u = __float_as_uint(f);
    unsigned int r = u + 0x7FFFu + ((u >> 16) & 1u);
    return (unsigned short)(r >> 16);
}

// A fragment: 8 shorts at p, 8 shorts at p+16 (16-bit A VGPR layout, K split by lane-half)
static __device__ __forceinline__ v16bf frag_gap(const unsigned short* p) {
    v8bf lo = *(const v8bf*)p;
    v8bf hi = *(const v8bf*)(p + 16);
    return __builtin_shufflevector(lo, hi, 0,1,2,3,4,5,6,7,8,9,10,11,12,13,14,15);
}
// B fragment: 8 shorts at p, 8 shorts at p+16 (column = lane&15, K split by lane-half along d of W)
static __device__ __forceinline__ v16bf frag_cont(const unsigned short* p) {
    v8bf lo = *(const v8bf*)p;
    v8bf hi = *(const v8bf*)(p + 16);
    return __builtin_shufflevector(lo, hi, 0,1,2,3,4,5,6,7,8,9,10,11,12,13,14,15);
}

// ---------------- prefix scan of chunk lens -> starts ----------------
__global__ void scan_starts_kernel(const int* __restrict__ lens, int* __restrict__ starts) {
    __shared__ int s[CN];
    const int b = blockIdx.x, t = threadIdx.x;
    const int v = lens[b * CN + t];
    s[t] = v;
    __syncthreads();
    for (int off = 1; off < CN; off <<= 1) {
        int w = (t >= off) ? s[t - off] : 0;
        __syncthreads();
        s[t] += w;
        __syncthreads();
    }
    starts[b * CN + t] = s[t] - v;   // exclusive prefix
}

// ---------------- dense_w (f32, row=e) -> bf16 ----------------
__global__ void wconv_kernel(const float* __restrict__ w, unsigned short* __restrict__ wbf) {
    int i = blockIdx.x * blockDim.x + threadIdx.x;
    if (i < D_DIM * D_DIM) wbf[i] = f32_to_bf16_rne(w[i]);
}

// ---------------- fused gather + attention + layernorm ----------------
__global__ __launch_bounds__(256) void chunk_attn_kernel(
    const float* __restrict__ tokens,          // [B,512,768]
    const int*   __restrict__ lens,            // [B,128]
    const int*   __restrict__ starts,          // [B,128]
    const unsigned short* __restrict__ wbf,    // [768,768] bf16 (row = e)
    const float* __restrict__ dense_b,
    const float* __restrict__ attn_bias,
    const float* __restrict__ query,
    const float* __restrict__ ln_w,
    const float* __restrict__ ln_b,
    float* __restrict__ chunk_feature)         // [B,128,768]
{
    __shared__ unsigned short sXbf[MROWS][XPADB];
    __shared__ float          sXf [MROWS][XPADF];
    __shared__ int            sRow[MROWS];
    __shared__ float          sLogit[MROWS];
    __shared__ float          sA[MROWS];
    __shared__ float          sSum[CP], sVar[CP];

    const int tid  = threadIdx.x;
    const int lane = tid & 31;
    const int wave = tid >> 5;
    const int wg   = blockIdx.x;              // 0..2047
    const int gc0  = wg * CP;                 // first global chunk (both chunks same batch)
    const int b    = gc0 / CN;

    // per-row gather metadata (row -> source token index, -1 if padded)
    if (tid < MROWS) {
        const int r   = tid;
        const int gc  = gc0 + (r >> 3);
        const int j   = r & (LC - 1);
        const int len = lens[gc];
        const int lim = len < LC ? len : LC;
        int src = starts[gc] + j;
        src = src < 0 ? 0 : (src > SEQ - 1 ? SEQ - 1 : src);
        sRow[r]   = (j < lim) ? src : -1;
        sLogit[r] = 0.0f;
    }
    if (tid < CP) { sSum[tid] = 0.0f; sVar[tid] = 0.0f; }
    __syncthreads();

    // gather X rows into LDS (fp32 for the weighted sum, bf16 for WMMA A)
    const float* tokB = tokens + (size_t)b * SEQ * D_DIM;
    for (int idx = tid; idx < MROWS * D_DIM; idx += 256) {
        const int r = idx / D_DIM;
        const int d = idx - r * D_DIM;
        const int src = sRow[r];
        const float v = (src >= 0) ? tokB[(size_t)src * D_DIM + d] : 0.0f;
        sXf[r][d]  = v;
        sXbf[r][d] = f32_to_bf16_rne(v);
    }
    __syncthreads();

    // ---- matmul h = tanh(X Wt + bias), fused logit = h . q ----
    float logacc[8];
    #pragma unroll
    for (int i = 0; i < 8; ++i) logacc[i] = 0.0f;

    const int m     = lane & 15;      // A row / B column / D column
    const int khalf = lane >> 4;      // lane-half K split
    const unsigned short* arow = &sXbf[m][khalf * 8];

    // each wave: 3 passes x 2 e-tiles (one A fragment feeds two independent WMMA chains)
    for (int it = 0; it < 3; ++it) {
        const int et0 = 2 * wave + 16 * it;
        const int e0  = et0 * 16 + m;
        const int e1  = e0 + 16;
        const unsigned short* wrow0 = wbf + (size_t)e0 * D_DIM + khalf * 8;
        const unsigned short* wrow1 = wbf + (size_t)e1 * D_DIM + khalf * 8;

        // prefetch next pass's W rows (outside the hot loop)
        if (it < 2) {
            __builtin_prefetch(wrow0 + 256 * D_DIM, 0, 1);
            __builtin_prefetch(wrow1 + 256 * D_DIM, 0, 1);
        }

        v8f acc0 = {}, acc1 = {};
        // software-pipelined: fragments for step kt+1 load while step kt's WMMAs run
        v16bf a_cur  = frag_gap (arow);
        v16bf b0_cur = frag_cont(wrow0);
        v16bf b1_cur = frag_cont(wrow1);
        #pragma unroll
        for (int kt = 1; kt < KSTEPS; ++kt) {
            v16bf a_n  = frag_gap (arow  + kt * 32);
            v16bf b0_n = frag_cont(wrow0 + kt * 32);
            v16bf b1_n = frag_cont(wrow1 + kt * 32);
            acc0 = __builtin_amdgcn_wmma_f32_16x16x32_bf16(
                false, a_cur, false, b0_cur, (short)0, acc0, false, false);
            acc1 = __builtin_amdgcn_wmma_f32_16x16x32_bf16(
                false, a_cur, false, b1_cur, (short)0, acc1, false, false);
            a_cur = a_n; b0_cur = b0_n; b1_cur = b1_n;
        }
        acc0 = __builtin_amdgcn_wmma_f32_16x16x32_bf16(
            false, a_cur, false, b0_cur, (short)0, acc0, false, false);
        acc1 = __builtin_amdgcn_wmma_f32_16x16x32_bf16(
            false, a_cur, false, b1_cur, (short)0, acc1, false, false);

        // consume both h tiles in registers: bias + tanh + dot with query
        const float bias0 = dense_b[e0] + attn_bias[e0];
        const float bias1 = dense_b[e1] + attn_bias[e1];
        const float q0 = query[e0];
        const float q1 = query[e1];
        #pragma unroll
        for (int v = 0; v < 8; ++v) {
            logacc[v] += tanhf(acc0[v] + bias0) * q0
                       + tanhf(acc1[v] + bias1) * q1;   // row M = v + 8*khalf
        }
    }

    // reduce logits over the 16 columns held by each lane-half, then across waves
    #pragma unroll
    for (int v = 0; v < 8; ++v) {
        float s = logacc[v];
        s += __shfl_xor(s, 8, 32);
        s += __shfl_xor(s, 4, 32);
        s += __shfl_xor(s, 2, 32);
        s += __shfl_xor(s, 1, 32);
        if (m == 0) atomicAdd(&sLogit[v + 8 * khalf], s);
    }
    __syncthreads();

    // softmax over the 8 positions of each chunk (zero rows handled naturally)
    if (tid < CP) {
        const int base = tid * LC;
        float mx = -1e30f;
        #pragma unroll
        for (int l = 0; l < LC; ++l) mx = fmaxf(mx, sLogit[base + l]);
        float ex[LC], sum = 0.0f;
        #pragma unroll
        for (int l = 0; l < LC; ++l) { ex[l] = __expf(sLogit[base + l] - mx); sum += ex[l]; }
        const float inv = 1.0f / sum;
        #pragma unroll
        for (int l = 0; l < LC; ++l) sA[base + l] = ex[l] * inv;
    }
    __syncthreads();

    // weighted sum over fp32 X + layernorm, 3 d-values per thread per chunk
    for (int c = 0; c < CP; ++c) {
        float o[3];
        float lsum = 0.0f;
        #pragma unroll
        for (int p = 0; p < 3; ++p) {
            const int d = tid + p * 256;
            float a = 0.0f;
            #pragma unroll
            for (int l = 0; l < LC; ++l) a += sA[c * LC + l] * sXf[c * LC + l][d];
            o[p] = a;
            lsum += a;
        }
        lsum += __shfl_xor(lsum, 16, 32);
        lsum += __shfl_xor(lsum,  8, 32);
        lsum += __shfl_xor(lsum,  4, 32);
        lsum += __shfl_xor(lsum,  2, 32);
        lsum += __shfl_xor(lsum,  1, 32);
        if (lane == 0) atomicAdd(&sSum[c], lsum);
        __syncthreads();
        const float mean = sSum[c] * (1.0f / D_DIM);
        float lvar = 0.0f;
        #pragma unroll
        for (int p = 0; p < 3; ++p) { const float t = o[p] - mean; lvar += t * t; }
        lvar += __shfl_xor(lvar, 16, 32);
        lvar += __shfl_xor(lvar,  8, 32);
        lvar += __shfl_xor(lvar,  4, 32);
        lvar += __shfl_xor(lvar,  2, 32);
        lvar += __shfl_xor(lvar,  1, 32);
        if (lane == 0) atomicAdd(&sVar[c], lvar);
        __syncthreads();
        const float rstd = rsqrtf(sVar[c] * (1.0f / D_DIM) + 1e-12f);
        float* outp = chunk_feature + (size_t)(gc0 + c) * D_DIM;
        #pragma unroll
        for (int p = 0; p < 3; ++p) {
            const int d = tid + p * 256;
            outp[d] = ln_w[d] * (o[p] - mean) * rstd + ln_b[d];
        }
    }
}

// ---------------- sentence embedding: max over 128 chunks ----------------
__global__ void sent_max_kernel(const float* __restrict__ cf, float* __restrict__ sent) {
    const int b = blockIdx.x;
    const int d = blockIdx.y * 256 + threadIdx.x;
    const float* p = cf + (size_t)b * CN * D_DIM + d;
    float m = p[0];
    for (int c = 1; c < CN; ++c) m = fmaxf(m, p[(size_t)c * D_DIM]);
    sent[b * D_DIM + d] = m;
}

extern "C" void kernel_launch(void* const* d_in, const int* in_sizes, int n_in,
                              void* d_out, int out_size, void* d_ws, size_t ws_size,
                              hipStream_t stream) {
    const float* tokens    = (const float*)d_in[0];
    const int*   lens      = (const int*)  d_in[1];
    const float* dense_w   = (const float*)d_in[2];
    const float* dense_b   = (const float*)d_in[3];
    const float* attn_bias = (const float*)d_in[4];
    const float* query     = (const float*)d_in[5];
    const float* ln_w      = (const float*)d_in[6];
    const float* ln_b      = (const float*)d_in[7];

    float* chunk_feature = (float*)d_out;                              // [32,128,768]
    float* sentence      = (float*)d_out + (size_t)BATCH * CN * D_DIM; // [32,768]

    unsigned short* wbf = (unsigned short*)d_ws;                       // 768*768 bf16
    int* starts = (int*)((char*)d_ws + (size_t)D_DIM * D_DIM * sizeof(unsigned short));

    scan_starts_kernel<<<BATCH, CN, 0, stream>>>(lens, starts);
    wconv_kernel<<<(D_DIM * D_DIM + 255) / 256, 256, 0, stream>>>(dense_w, wbf);
    chunk_attn_kernel<<<NCHUNKS / CP, 256, 0, stream>>>(
        tokens, lens, starts, wbf, dense_b, attn_bias, query, ln_w, ln_b, chunk_feature);
    sent_max_kernel<<<dim3(BATCH, D_DIM / 256), 256, 0, stream>>>(chunk_feature, sentence);
}
